// SelfAttention_v2_64132451663995
// MI455X (gfx1250) — matrix operand order repaired and checked by reading, not compiled, not run on hardware
//
#include <hip/hip_runtime.h>

// ---------------------------------------------------------------------------
// Self-attention (B=4, S=2048, D=1024), softmax-then-tril-mask semantics.
// All matmuls via v_wmma_f32_16x16x32_bf16 (CDNA5 wave32 WMMA).
// Staging uses gfx1250 async global->LDS DMA (ASYNCcnt) + double buffering,
// unconditional prefetch, K-loop unrolled x2 so buffer parity is static.
// ---------------------------------------------------------------------------

typedef __attribute__((ext_vector_type(16))) __bf16 v16bf;
typedef __attribute__((ext_vector_type(8)))  __bf16 v8bf;
typedef __attribute__((ext_vector_type(8)))  float  v8f;
using u16 = unsigned short;

static constexpr int B_  = 4;
static constexpr int S_  = 2048;
static constexpr int D_  = 1024;

// round-to-nearest-even f32 -> bf16 (bit pattern as u16)
__device__ __forceinline__ u16 f2bf(float f) {
    unsigned u = __builtin_bit_cast(unsigned, f);
    u += 0x7FFFu + ((u >> 16) & 1u);
    return (u16)(u >> 16);
}

union V16 { v16bf v; v8bf h[2]; };

// 16-byte async DMA: global -> LDS, tracked by ASYNCcnt.
// LDS aperture maps flat addr[31:0] directly to the LDS byte address.
__device__ __forceinline__ void async_cp16(const u16* g, u16* l) {
    unsigned           lds32 = (unsigned)(unsigned long long)(uintptr_t)l;
    unsigned long long ga    = (unsigned long long)(uintptr_t)g;
    asm volatile("global_load_async_to_lds_b128 %0, %1, off"
                 :: "v"(lds32), "v"(ga) : "memory");
}

// ---------------------------------------------------------------------------
// elementwise f32 -> bf16 (n must be multiple of 8)
// ---------------------------------------------------------------------------
__global__ void cvt_bf16_kernel(const float* __restrict__ src,
                                u16* __restrict__ dst, int n) {
    int i = (blockIdx.x * 256 + threadIdx.x) * 8;
    if (i >= n) return;
    float4 a = *(const float4*)(src + i);
    float4 b = *(const float4*)(src + i + 4);
    u16 o[8] = { f2bf(a.x), f2bf(a.y), f2bf(a.z), f2bf(a.w),
                 f2bf(b.x), f2bf(b.y), f2bf(b.z), f2bf(b.w) };
    *(uint4*)(dst + i) = *(const uint4*)o;
}

// ---------------------------------------------------------------------------
// W [D][D] f32  ->  Wt [D][D] bf16, transposed (Wt[e][d] = W[d][e])
// ---------------------------------------------------------------------------
__global__ void wtrans_kernel(const float* __restrict__ W,
                              u16* __restrict__ Wt) {
    int e = blockIdx.x * 16 + threadIdx.x;
    int d = blockIdx.y * 16 + threadIdx.y;
    Wt[(size_t)e * D_ + d] = f2bf(W[(size_t)d * D_ + e]);
}

// ---------------------------------------------------------------------------
// GEMM: C[M][N] = scale * A[M][K] * B[N][K]^T   (A,B bf16 row-major)
// OMODE: 0 = bf16 store, 1 = f32 store, 2 = bf16 transposed store Ct[n][m]
// Block tile 128x128, 8 waves (256 thr), wave tile 32x64, K-step 32.
// Double-buffered LDS, async global->LDS staging (4 DMAs/thread/step).
// The K-loop always prefetches the next 32-K slab; on the final step this
// reads <=64B past the logical tile, which stays inside the workspace for
// every launch below (adjacent regions) and is never consumed.
// M,N multiples of 128; K a multiple of 64.
// ---------------------------------------------------------------------------
static constexpr int LDS_STRIDE = 40;   // 32 bf16 + 8 pad (80B row, 16B aligned)

template <int OMODE>
__global__ __launch_bounds__(256)
void gemm_bf16_kernel(const u16* __restrict__ A, const u16* __restrict__ Bm,
                      void* __restrict__ C,
                      int K, int lda, int ldb, int ldc,
                      long batchA, long batchB, long batchC, float scale) {
    __shared__ u16 sA[2][128 * LDS_STRIDE];
    __shared__ u16 sB[2][128 * LDS_STRIDE];

    const int z = blockIdx.z;
    A  += (size_t)z * batchA;
    Bm += (size_t)z * batchB;

    const int m0 = blockIdx.y * 128;
    const int n0 = blockIdx.x * 128;

    const int t    = threadIdx.x;
    const int lane = t & 31;           // wave32
    const int w    = t >> 5;           // 8 waves
    const int wm   = w & 3;            // 4 waves along M
    const int wn   = w >> 2;           // 2 waves along N
    const int r    = lane & 15;
    const int hi   = lane >> 4;

    // this thread's two (row, k-chunk) staging slots
    const int row0 = t >> 2,         kc0 = (t & 3) * 8;
    const int row1 = (t + 256) >> 2, kc1 = ((t + 256) & 3) * 8;

    const u16* gA0 = &A[(size_t)(m0 + row0) * lda + kc0];
    const u16* gA1 = &A[(size_t)(m0 + row1) * lda + kc1];
    const u16* gB0 = &Bm[(size_t)(n0 + row0) * ldb + kc0];
    const u16* gB1 = &Bm[(size_t)(n0 + row1) * ldb + kc1];

    auto stage = [&](int k0, int buf) {
        async_cp16(gA0 + k0, &sA[buf][row0 * LDS_STRIDE + kc0]);
        async_cp16(gA1 + k0, &sA[buf][row1 * LDS_STRIDE + kc1]);
        async_cp16(gB0 + k0, &sB[buf][row0 * LDS_STRIDE + kc0]);
        async_cp16(gB1 + k0, &sB[buf][row1 * LDS_STRIDE + kc1]);
    };

    v8f acc[2][4] = {};

    auto body = [&](int k0, int buf) {
        __syncthreads();              // all waves done reading buf^1
        stage(k0 + 32, buf ^ 1);      // unconditional prefetch of next slab
        asm volatile("s_wait_asynccnt 4" ::: "memory");  // this slab landed (mine)
        __syncthreads();              // everyone's stores for this slab landed

        // ---- build fragments (ISA §7.12.2 bf16 layouts) ----
        V16 av[2];
        #pragma unroll
        for (int i = 0; i < 2; ++i) {
            int row = wm * 32 + i * 16 + r;
            av[i].h[0] = *(const v8bf*)&sA[buf][row * LDS_STRIDE + hi * 8];
            av[i].h[1] = *(const v8bf*)&sA[buf][row * LDS_STRIDE + 16 + hi * 8];
        }
        V16 bv[4];
        #pragma unroll
        for (int j = 0; j < 4; ++j) {
            int row = wn * 64 + j * 16 + r;                  // N column
            bv[j].h[0] = *(const v8bf*)&sB[buf][row * LDS_STRIDE + hi * 8];
            bv[j].h[1] = *(const v8bf*)&sB[buf][row * LDS_STRIDE + 16 + hi * 8];
        }

        // ---- 8 WMMAs per wave per K-step ----
        #pragma unroll
        for (int i = 0; i < 2; ++i)
            #pragma unroll
            for (int j = 0; j < 4; ++j)
                acc[i][j] = __builtin_amdgcn_wmma_f32_16x16x32_bf16(
                    false, av[i].v, false, bv[j].v, (short)0, acc[i][j],
                    false, false);
    };

    stage(0, 0);                       // prologue: slab 0 in flight
    for (int k0 = 0; k0 < K; k0 += 64) {   // K/32 even -> exact 2x unroll
        body(k0, 0);
        body(k0 + 32, 1);
    }

    // ---- epilogue: D VGPR e holds (M = e + 8*hi, N = lane&15) ----
    #pragma unroll
    for (int i = 0; i < 2; ++i)
        #pragma unroll
        for (int j = 0; j < 4; ++j)
            #pragma unroll
            for (int e = 0; e < 8; ++e) {
                int m = m0 + wm * 32 + i * 16 + e + hi * 8;
                int n = n0 + wn * 64 + j * 16 + r;
                float v = acc[i][j][e] * scale;
                if (OMODE == 1) {
                    float* Cp = (float*)C + (size_t)z * batchC;
                    Cp[(size_t)m * ldc + n] = v;
                } else if (OMODE == 0) {
                    u16* Cp = (u16*)C + (size_t)z * batchC;
                    Cp[(size_t)m * ldc + n] = f2bf(v);
                } else {                       // transposed bf16: Ct[n][m]
                    u16* Cp = (u16*)C + (size_t)z * batchC;
                    Cp[(size_t)n * ldc + m] = f2bf(v);
                }
            }
}

// ---------------------------------------------------------------------------
// Row softmax over all 2048 keys, then zero mask for k > q (no renorm).
// Reads fp32 row, writes bf16 probabilities in place over the row start.
// One block (256 thr) per (b, q) row; 8 elements per thread.
// ---------------------------------------------------------------------------
__global__ __launch_bounds__(256)
void softmax_mask_kernel(float* __restrict__ scores) {
    const int row = blockIdx.x;            // 0 .. B*S-1
    const int q   = row & (S_ - 1);
    float* rp = scores + (size_t)row * S_;

    const int t = threadIdx.x;
    float v[8];
    {
        float4 a = *(const float4*)(rp + t * 8);
        float4 b = *(const float4*)(rp + t * 8 + 4);
        v[0]=a.x; v[1]=a.y; v[2]=a.z; v[3]=a.w;
        v[4]=b.x; v[5]=b.y; v[6]=b.z; v[7]=b.w;
    }

    __shared__ float red[256];
    float mx = v[0];
    #pragma unroll
    for (int i = 1; i < 8; ++i) mx = fmaxf(mx, v[i]);
    red[t] = mx;
    __syncthreads();
    for (int s = 128; s > 0; s >>= 1) {
        if (t < s) red[t] = fmaxf(red[t], red[t + s]);
        __syncthreads();
    }
    mx = red[0];
    __syncthreads();

    float e[8], sum = 0.f;
    #pragma unroll
    for (int i = 0; i < 8; ++i) { e[i] = __expf(v[i] - mx); sum += e[i]; }
    red[t] = sum;
    __syncthreads();
    for (int s = 128; s > 0; s >>= 1) {
        if (t < s) red[t] += red[t + s];
        __syncthreads();
    }
    const float inv = 1.0f / red[0];

    // masked bf16 probabilities written over the start of this fp32 row
    u16 o[8];
    #pragma unroll
    for (int i = 0; i < 8; ++i) {
        int k = t * 8 + i;
        o[i] = f2bf((k <= q) ? e[i] * inv : 0.0f);
    }
    *(uint4*)((u16*)rp + t * 8) = *(const uint4*)o;
}

// ---------------------------------------------------------------------------
// Orchestration. Workspace layout (bytes):
//   [0,16M)   Qb   bf16  [B][S][D]   (pre-scaled by 1/32)
//   [16,32M)  Kb   bf16  [B][S][D]
//   [32,48M)  Vt   bf16  [B][D][S]   (transposed for context GEMM)
//   [48,64M)  Xb   bf16  [B][S][D]
//   [64,70M)  Wqt/Wkt/Wvt bf16 [D][D] transposed (2MB each)
//   [72,136M) scores fp32 [B][S][S]  (bf16 probs written in place per row)
// ---------------------------------------------------------------------------
extern "C" void kernel_launch(void* const* d_in, const int* in_sizes, int n_in,
                              void* d_out, int out_size, void* d_ws, size_t ws_size,
                              hipStream_t stream) {
    (void)in_sizes; (void)n_in; (void)out_size; (void)ws_size;

    const float* x  = (const float*)d_in[0];
    const float* Wq = (const float*)d_in[1];
    const float* Wk = (const float*)d_in[2];
    const float* Wv = (const float*)d_in[3];

    const size_t MBY = 1u << 20;
    char* ws = (char*)d_ws;
    u16*   Qb  = (u16*)(ws + 0 * MBY);
    u16*   Kb  = (u16*)(ws + 16 * MBY);
    u16*   Vt  = (u16*)(ws + 32 * MBY);
    u16*   Xb  = (u16*)(ws + 48 * MBY);
    u16*   Wqt = (u16*)(ws + 64 * MBY);
    u16*   Wkt = (u16*)(ws + 66 * MBY);
    u16*   Wvt = (u16*)(ws + 68 * MBY);
    float* SC  = (float*)(ws + 72 * MBY);

    const long SD  = (long)S_ * D_;      // 2M elems
    const long SS  = (long)S_ * S_;      // 4M elems

    // 1) x -> bf16
    {
        int n = B_ * S_ * D_;
        cvt_bf16_kernel<<<dim3((n / 8 + 255) / 256), dim3(256), 0, stream>>>(x, Xb, n);
    }
    // 2) W -> bf16 transposed
    {
        dim3 g(D_ / 16, D_ / 16), b(16, 16);
        wtrans_kernel<<<g, b, 0, stream>>>(Wq, Wqt);
        wtrans_kernel<<<g, b, 0, stream>>>(Wk, Wkt);
        wtrans_kernel<<<g, b, 0, stream>>>(Wv, Wvt);
    }
    // 3) projections: Q (scaled 1/sqrt(1024)=1/32), K, Vt (transposed store)
    {
        dim3 g(D_ / 128, S_ / 128, B_), b(256);
        gemm_bf16_kernel<0><<<g, b, 0, stream>>>(Xb, Wqt, Qb,
            D_, D_, D_, D_, SD, 0, SD, 0.03125f);
        gemm_bf16_kernel<0><<<g, b, 0, stream>>>(Xb, Wkt, Kb,
            D_, D_, D_, D_, SD, 0, SD, 1.0f);
        gemm_bf16_kernel<2><<<g, b, 0, stream>>>(Xb, Wvt, Vt,
            D_, D_, D_, /*ldc=M stride*/ S_, SD, 0, SD, 1.0f);
    }
    // 4) scores = Q @ K^T  (fp32)
    {
        dim3 g(S_ / 128, S_ / 128, B_), b(256);
        gemm_bf16_kernel<1><<<g, b, 0, stream>>>(Qb, Kb, SC,
            D_, D_, D_, S_, SD, SD, SS, 1.0f);
    }
    // 5) softmax over all keys + causal zero-mask; bf16 P in place
    softmax_mask_kernel<<<dim3(B_ * S_), dim3(256), 0, stream>>>(SC);

    // 6) context = P @ V   (A = bf16 P rows, lda = 2*S ushorts; B = Vt [e][s])
    {
        dim3 g(D_ / 128, S_ / 128, B_), b(256);
        gemm_bf16_kernel<1><<<g, b, 0, stream>>>((const u16*)SC, Vt, d_out,
            S_, /*lda*/ 2 * S_, /*ldb*/ S_, /*ldc*/ D_,
            /*batchA ushorts*/ 2L * S_ * S_, /*batchB*/ (long)D_ * S_,
            /*batchC*/ SD, 1.0f);
    }
}
